// MultiHeadHierarchicalAttention_52458730554059
// MI455X (gfx1250) — matrix operand order repaired and checked by reading, not compiled, not run on hardware
//
#include <hip/hip_runtime.h>

typedef __attribute__((ext_vector_type(16))) __bf16 v16bf;
typedef __attribute__((ext_vector_type(8)))  __bf16 v8bf;
typedef __attribute__((ext_vector_type(8)))  float  v8f;
typedef __attribute__((ext_vector_type(4)))  float  v4f;

#define DEV static __device__ __forceinline__

// ---- problem constants ----
#define CB    2
#define CLQ   512
#define CNB   24
#define CNBP  32      // NB padded
#define CNT   256
#define CNTOP 100
#define CTP   128     // NTOP padded
#define CDM   512
#define CH    8
#define CDK   64
#define CSCALE 0.125f // 64^-0.5

// ---------------- WMMA helpers ----------------
DEV v8f wmma_bf(v16bf a, v16bf b, v8f c) {
  return __builtin_amdgcn_wmma_f32_16x16x32_bf16(false, a, false, b, (short)0, c,
                                                 false, false);
}

DEV v16bf cat8(v8bf lo, v8bf hi) {
  return __builtin_shufflevector(lo, hi, 0, 1, 2, 3, 4, 5, 6, 7, 8, 9, 10, 11, 12,
                                 13, 14, 15);
}

// A fragment 16x32: lane holds row m0+(lane&15); lanes0-15 K{0..7,16..23},
// lanes16-31 K{8..15,24..31}. Two contiguous 16B loads per lane.
DEV v16bf load_a_bf(const __bf16* src, int ld, int m0, int k0) {
  int lane = threadIdx.x & 31;
  const __bf16* p =
      src + (size_t)(m0 + (lane & 15)) * ld + k0 + ((lane & 16) ? 8 : 0);
  v8bf lo = *(const v8bf*)p;
  v8bf hi = *(const v8bf*)(p + 16);
  return cat8(lo, hi);
}

// B fragment 32x16 where B[k][n] = srcT[n][k] (srcT row-major [N][K]).
DEV v16bf load_bt_bf(const __bf16* srcT, int ld, int n0, int k0) {
  int lane = threadIdx.x & 31;
  const __bf16* p =
      srcT + (size_t)(n0 + (lane & 15)) * ld + k0 + ((lane & 16) ? 8 : 0);
  v8bf lo = *(const v8bf*)p;
  v8bf hi = *(const v8bf*)(p + 16);
  return cat8(lo, hi);
}

// A fragment from f32 LDS tile, converting to bf16.
DEV v16bf load_a_f32(const float* src, int ld, int m0, int k0) {
  int lane = threadIdx.x & 31;
  const float* p =
      src + (size_t)(m0 + (lane & 15)) * ld + k0 + ((lane & 16) ? 8 : 0);
  v4f a = *(const v4f*)p;
  v4f b = *(const v4f*)(p + 4);
  v4f c = *(const v4f*)(p + 16);
  v4f d = *(const v4f*)(p + 20);
  v16bf f;
#pragma unroll
  for (int i = 0; i < 4; ++i) {
    f[i]      = (__bf16)a[i];
    f[4 + i]  = (__bf16)b[i];
    f[8 + i]  = (__bf16)c[i];
    f[12 + i] = (__bf16)d[i];
  }
  return f;
}

// C/D layout: VGPR r: lanes0-15 -> (M=r,N=lane), lanes16-31 -> (M=8+r,N=lane-16)
DEV void store_c_f32(float* dst, int ld, int m0, int n0, v8f c) {
  int lane = threadIdx.x & 31;
  int n = n0 + (lane & 15);
  int m = m0 + ((lane & 16) ? 8 : 0);
#pragma unroll
  for (int r = 0; r < 8; ++r) dst[(size_t)(m + r) * ld + n] = c[r];
}

DEV void store_c_bf16(__bf16* dst, int ld, int m0, int n0, v8f c) {
  int lane = threadIdx.x & 31;
  int n = n0 + (lane & 15);
  int m = m0 + ((lane & 16) ? 8 : 0);
#pragma unroll
  for (int r = 0; r < 8; ++r) dst[(size_t)(m + r) * ld + n] = (__bf16)c[r];
}

// Transposed store: CT[n][m..m+7], one contiguous 16B store per lane.
DEV void store_ct_bf16(__bf16* dstT, int ldT, int m0, int n0, v8f c) {
  int lane = threadIdx.x & 31;
  int n = n0 + (lane & 15);
  int m = m0 + ((lane & 16) ? 8 : 0);
  v8bf t;
#pragma unroll
  for (int r = 0; r < 8; ++r) t[r] = (__bf16)c[r];
  *(v8bf*)(dstT + (size_t)n * ldT + m) = t;
}

// ---------------- utility kernels ----------------
__global__ void k_conv(const float* __restrict__ src, __bf16* __restrict__ dst,
                       long n) {
  long i = (long)blockIdx.x * blockDim.x + threadIdx.x;
  if (i < n) dst[i] = (__bf16)src[i];
}

__global__ void k_conv_pad(const float* __restrict__ src, __bf16* __restrict__ dst,
                           int srows, int drows, int cols, long total) {
  long i = (long)blockIdx.x * blockDim.x + threadIdx.x;
  if (i >= total) return;
  int c = (int)(i % cols);
  long rb = i / cols;
  int r = (int)(rb % drows);
  long b = rb / drows;
  dst[i] = (r < srows) ? (__bf16)src[((size_t)b * srows + r) * cols + c]
                       : (__bf16)0.0f;
}

__global__ void k_conv_T(const float* __restrict__ src, __bf16* __restrict__ dst,
                         int K, int N) {
  long i = (long)blockIdx.x * blockDim.x + threadIdx.x;
  if (i >= (long)K * N) return;
  int k = (int)(i / N);
  int n = (int)(i % N);
  dst[(size_t)n * K + k] = (__bf16)src[i];
}

// ---------------- GEMM: C = A[M,K] * W[K,N] + bias ----------------
// WT is [N][K]. 2x2 register blocking, all dims multiples of 32. 4 waves/block.
__global__ void k_gemm_bf16(const __bf16* __restrict__ A, size_t sA, int lda,
                            const __bf16* __restrict__ WT, size_t sW, int ldwT,
                            const float* __restrict__ bias,
                            float* __restrict__ Cf, __bf16* __restrict__ Cb,
                            __bf16* __restrict__ CbT, size_t sC, int ldc,
                            size_t sCT, int ldt, int M, int N, int K) {
  int batch = blockIdx.y;
  int tiles_n = N >> 5;
  int tiles_m = M >> 5;
  int tile = blockIdx.x * (blockDim.x >> 5) + (threadIdx.x >> 5);
  if (tile >= tiles_m * tiles_n) return;
  int m0 = (tile / tiles_n) << 5;
  int n0 = (tile % tiles_n) << 5;
  const __bf16* Ab = A + (size_t)batch * sA;
  const __bf16* Wb = WT + (size_t)batch * sW;
  v8f a00 = {}, a01 = {}, a10 = {}, a11 = {};
  for (int k0 = 0; k0 < K; k0 += 32) {
    v16bf fa0 = load_a_bf(Ab, lda, m0, k0);
    v16bf fa1 = load_a_bf(Ab, lda, m0 + 16, k0);
    v16bf fb0 = load_bt_bf(Wb, ldwT, n0, k0);
    v16bf fb1 = load_bt_bf(Wb, ldwT, n0 + 16, k0);
    a00 = wmma_bf(fa0, fb0, a00);
    a01 = wmma_bf(fa0, fb1, a01);
    a10 = wmma_bf(fa1, fb0, a10);
    a11 = wmma_bf(fa1, fb1, a11);
  }
  if (bias) {
    int nn = threadIdx.x & 15;
    float b0 = bias[n0 + nn];
    float b1 = bias[n0 + 16 + nn];
#pragma unroll
    for (int r = 0; r < 8; ++r) {
      a00[r] += b0; a10[r] += b0;
      a01[r] += b1; a11[r] += b1;
    }
  }
  if (Cf) {
    float* C = Cf + (size_t)batch * sC;
    store_c_f32(C, ldc, m0, n0, a00);
    store_c_f32(C, ldc, m0, n0 + 16, a01);
    store_c_f32(C, ldc, m0 + 16, n0, a10);
    store_c_f32(C, ldc, m0 + 16, n0 + 16, a11);
  }
  if (Cb) {
    __bf16* C = Cb + (size_t)batch * sC;
    store_c_bf16(C, ldc, m0, n0, a00);
    store_c_bf16(C, ldc, m0, n0 + 16, a01);
    store_c_bf16(C, ldc, m0 + 16, n0, a10);
    store_c_bf16(C, ldc, m0 + 16, n0 + 16, a11);
  }
  if (CbT) {
    __bf16* C = CbT + (size_t)batch * sCT;
    store_ct_bf16(C, ldt, m0, n0, a00);
    store_ct_bf16(C, ldt, m0, n0 + 16, a01);
    store_ct_bf16(C, ldt, m0 + 16, n0, a10);
    store_ct_bf16(C, ldt, m0 + 16, n0 + 16, a11);
  }
}

// ---------------- attention kernels (1 wave = 16 q rows) ----------------

// Topic attention: probs over NTOP (padded TP) + context via v_t^T.
__global__ void k_topic_attn(const __bf16* __restrict__ q_t,
                             const __bf16* __restrict__ k_t,
                             const __bf16* __restrict__ vtT,
                             const float* __restrict__ tbias,
                             __bf16* __restrict__ attn_t_bf,
                             __bf16* __restrict__ ctx_t_bf) {
  int id = blockIdx.x;
  int qt = id & 31; id >>= 5;
  int h = id % CH;
  int bb = id / CH;
  int q0 = qt * 16;
  int lane = threadIdx.x & 31;
  __shared__ float S[16][CTP];

  const __bf16* Aq = q_t + (size_t)bb * CLQ * CDM + h * CDK;
  const __bf16* Kt = k_t + (size_t)bb * CTP * CDM + h * CDK;
  const __bf16* Vt = vtT + ((size_t)bb * CDM + h * CDK) * CTP;
  const float* tb = tbias + (size_t)((bb * CH + h) * CLQ + q0) * CNTOP;

  v16bf aq0 = load_a_bf(Aq, CDM, q0, 0);
  v16bf aq1 = load_a_bf(Aq, CDM, q0, 32);

  int nn = lane & 15;
  int mo = (lane & 16) ? 8 : 0;
  for (int tt = 0; tt < CTP / 16; ++tt) {
    v8f acc = {};
    acc = wmma_bf(aq0, load_bt_bf(Kt, CDM, tt * 16, 0), acc);
    acc = wmma_bf(aq1, load_bt_bf(Kt, CDM, tt * 16, 32), acc);
    int t = tt * 16 + nn;
#pragma unroll
    for (int r = 0; r < 8; ++r) {
      int row = mo + r;
      S[row][t] =
          (t < CNTOP) ? acc[r] * CSCALE + tb[(size_t)row * CNTOP + t] : -1e30f;
    }
  }
  __syncthreads();
  {
    // 2 lanes per row: lane handles half [half*64, half*64+64)
    int row = lane & 15;
    int half = lane >> 4;
    int t0 = half * (CTP / 2);
    float mx = -1e30f;
    for (int t = t0; t < t0 + CTP / 2; ++t)
      if (t < CNTOP) mx = fmaxf(mx, S[row][t]);
    mx = fmaxf(mx, __shfl_xor(mx, 16));
    float sum = 0.f;
    for (int t = t0; t < t0 + CTP / 2; ++t) {
      float e = (t < CNTOP) ? __expf(S[row][t] - mx) : 0.0f;
      S[row][t] = e;
      sum += e;
    }
    sum += __shfl_xor(sum, 16);
    float inv = 1.0f / sum;
    __bf16* ao = attn_t_bf + (size_t)((bb * CH + h) * CLQ + q0 + row) * CTP;
    for (int t = t0; t < t0 + CTP / 2; ++t) {
      float p = S[row][t] * inv;
      S[row][t] = p;
      ao[t] = (__bf16)p;
    }
  }
  __syncthreads();
  __bf16* Co = ctx_t_bf + (size_t)bb * CLQ * CDM + h * CDK;
  v8f acc[4] = {};
  for (int kt = 0; kt < CTP / 32; ++kt) {
    v16bf a = load_a_f32(&S[0][0], CTP, 0, kt * 32);
#pragma unroll
    for (int nt = 0; nt < 4; ++nt) {
      v16bf b = load_bt_bf(Vt, CTP, nt * 16, kt * 32);
      acc[nt] = wmma_bf(a, b, acc[nt]);
    }
  }
  for (int nt = 0; nt < 4; ++nt)
    store_c_bf16(Co, CDM, q0, nt * 16, acc[nt]);
}

// Topic -> paragraph attention probs, written transposed: apT[p][t].
__global__ void k_para_attn(const __bf16* __restrict__ q_pp,
                            const __bf16* __restrict__ k_pp,
                            const float* __restrict__ ptbias,
                            __bf16* __restrict__ apT) {
  int id = blockIdx.x;
  int tt = id % (CTP / 16); id /= (CTP / 16);
  int h = id % CH;
  int bb = id / CH;
  int t0 = tt * 16;
  int lane = threadIdx.x & 31;
  __shared__ float S[16][CNBP];

  const __bf16* Aq = q_pp + (size_t)bb * CTP * CDM + h * CDK;
  const __bf16* Kp = k_pp + (size_t)bb * CNBP * CDM + h * CDK;

  v16bf a0 = load_a_bf(Aq, CDM, t0, 0);
  v16bf a1 = load_a_bf(Aq, CDM, t0, 32);

  int nn = lane & 15;
  int mo = (lane & 16) ? 8 : 0;
  for (int pt = 0; pt < 2; ++pt) {
    v8f acc = {};
    acc = wmma_bf(a0, load_bt_bf(Kp, CDM, pt * 16, 0), acc);
    acc = wmma_bf(a1, load_bt_bf(Kp, CDM, pt * 16, 32), acc);
    int p = pt * 16 + nn;
#pragma unroll
    for (int r = 0; r < 8; ++r) {
      int row = mo + r;
      int t = t0 + row;
      S[row][p] = (t < CNTOP && p < CNB)
                      ? acc[r] * CSCALE +
                            ptbias[(size_t)((bb * CH + h) * CNB + p) * CNTOP + t]
                      : -1e30f;
    }
  }
  __syncthreads();
  if (lane < 16) {
    int row = lane;
    int t = t0 + row;
    __bf16* ao = apT + (size_t)(bb * CH + h) * CNBP * CTP + t;
    if (t < CNTOP) {
      float mx = -1e30f;
      for (int p = 0; p < CNB; ++p) mx = fmaxf(mx, S[row][p]);
      float sum = 0.f;
      for (int p = 0; p < CNB; ++p) {
        float e = __expf(S[row][p] - mx);
        S[row][p] = e;
        sum += e;
      }
      float inv = 1.0f / sum;
      for (int p = 0; p < CNB; ++p) ao[(size_t)p * CTP] = (__bf16)(S[row][p] * inv);
      for (int p = CNB; p < CNBP; ++p) ao[(size_t)p * CTP] = (__bf16)0.0f;
    } else {
      for (int p = 0; p < CNBP; ++p) ao[(size_t)p * CTP] = (__bf16)0.0f;
    }
  }
}

// Sentence attention with para-topic renormalization.
__global__ void k_sent_attn(const __bf16* __restrict__ q_sp,
                            const __bf16* __restrict__ k_spp,
                            const __bf16* __restrict__ vsppT,
                            const float* __restrict__ bias_s,
                            const float* __restrict__ attn_tp,
                            float* __restrict__ attn_sn,
                            __bf16* __restrict__ ctx_s_bf) {
  int id = blockIdx.x;
  int qt = id & 31; id >>= 5;
  int h = id % CH;
  int bb = id / CH;
  int q0 = qt * 16;
  int lane = threadIdx.x & 31;
  __shared__ float S[16][CNBP];

  const __bf16* Aq = q_sp + (size_t)bb * CLQ * CDM + h * CDK;
  const __bf16* Ks = k_spp + (size_t)bb * CNBP * CDM + h * CDK;
  const __bf16* Vs = vsppT + ((size_t)bb * CDM + h * CDK) * CNBP;
  const float* bs = bias_s + (size_t)((bb * CH + h) * CLQ + q0) * CNB;
  const float* tp = attn_tp + (size_t)((bb * CH + h) * CLQ + q0) * CNBP;
  float* sn = attn_sn + (size_t)((bb * CH + h) * CLQ + q0) * CNBP;

  v16bf a0 = load_a_bf(Aq, CDM, q0, 0);
  v16bf a1 = load_a_bf(Aq, CDM, q0, 32);

  int nn = lane & 15;
  int mo = (lane & 16) ? 8 : 0;
  for (int pt = 0; pt < 2; ++pt) {
    v8f acc = {};
    acc = wmma_bf(a0, load_bt_bf(Ks, CDM, pt * 16, 0), acc);
    acc = wmma_bf(a1, load_bt_bf(Ks, CDM, pt * 16, 32), acc);
    int n = pt * 16 + nn;
#pragma unroll
    for (int r = 0; r < 8; ++r) {
      int row = mo + r;
      S[row][n] =
          (n < CNB) ? acc[r] * CSCALE + bs[(size_t)row * CNB + n] : -1e30f;
    }
  }
  __syncthreads();
  if (lane < 16) {
    int row = lane;
    float mx = -1e30f;
    for (int n = 0; n < CNB; ++n) mx = fmaxf(mx, S[row][n]);
    float sum = 0.f;
    for (int n = 0; n < CNB; ++n) {
      float e = __expf(S[row][n] - mx);
      S[row][n] = e;
      sum += e;
    }
    float inv = 1.0f / sum;
    float sum2 = 0.f;
    for (int n = 0; n < CNB; ++n) {
      float p = S[row][n] * inv * tp[(size_t)row * CNBP + n];
      S[row][n] = p;
      sum2 += p;
    }
    float inv2 = 1.0f / (sum2 + 1e-9f);
    for (int n = 0; n < CNB; ++n) {
      float p = S[row][n] * inv2;
      S[row][n] = p;
      sn[(size_t)row * CNBP + n] = p;
    }
    for (int n = CNB; n < CNBP; ++n) {
      S[row][n] = 0.0f;
      sn[(size_t)row * CNBP + n] = 0.0f;
    }
  }
  __syncthreads();
  __bf16* Co = ctx_s_bf + (size_t)bb * CLQ * CDM + h * CDK;
  v16bf a = load_a_f32(&S[0][0], CNBP, 0, 0);
  for (int nt = 0; nt < 4; ++nt) {
    v8f acc = {};
    v16bf b = load_bt_bf(Vs, CNBP, nt * 16, 0);
    acc = wmma_bf(a, b, acc);
    store_c_bf16(Co, CDM, q0, nt * 16, acc);
  }
}

// Word attention: loops NB blocks; context accumulated in registers.
__global__ void k_word_attn(const __bf16* __restrict__ q_wp,
                            const __bf16* __restrict__ kw_p,
                            const __bf16* __restrict__ vwT,
                            const float* __restrict__ bias_w,
                            const float* __restrict__ attn_sn,
                            __bf16* __restrict__ ctx_w_bf) {
  int id = blockIdx.x;
  int qt = id & 31; id >>= 5;
  int h = id % CH;
  int bb = id / CH;
  int q0 = qt * 16;
  int lane = threadIdx.x & 31;
  __shared__ float S[16][CNT];

  const __bf16* Aq = q_wp + (size_t)bb * CLQ * CDM + h * CDK;
  v16bf a0 = load_a_bf(Aq, CDM, q0, 0);
  v16bf a1 = load_a_bf(Aq, CDM, q0, 32);

  v8f ctx[4] = {};
  int nn = lane & 15;
  int mo = (lane & 16) ? 8 : 0;
  int srow = lane & 15;     // softmax row
  int shalf = lane >> 4;    // softmax half
  int st0 = shalf * (CNT / 2);

  for (int n = 0; n < CNB; ++n) {
    const __bf16* Kp = kw_p + ((size_t)(bb * CNB + n) * CNT) * CDM + h * CDK;
    const __bf16* Vp = vwT + ((size_t)(bb * CNB + n) * CDM + h * CDK) * CNT;
    const float* bw =
        bias_w + ((((size_t)bb * CNB + n) * CH + h) * CLQ + q0) * CNT;

    for (int tt = 0; tt < CNT / 16; ++tt) {
      v8f acc = {};
      acc = wmma_bf(a0, load_bt_bf(Kp, CDM, tt * 16, 0), acc);
      acc = wmma_bf(a1, load_bt_bf(Kp, CDM, tt * 16, 32), acc);
      int t = tt * 16 + nn;
#pragma unroll
      for (int r = 0; r < 8; ++r) {
        int row = mo + r;
        S[row][t] = acc[r] * CSCALE + bw[(size_t)row * CNT + t];
      }
    }
    __syncthreads();
    {
      // 2 lanes per row, each handles 128 elements; combine via shfl_xor(16)
      float mx = -1e30f;
      for (int t = st0; t < st0 + CNT / 2; ++t) mx = fmaxf(mx, S[srow][t]);
      mx = fmaxf(mx, __shfl_xor(mx, 16));
      float sum = 0.f;
      for (int t = st0; t < st0 + CNT / 2; ++t) {
        float e = __expf(S[srow][t] - mx);
        S[srow][t] = e;
        sum += e;
      }
      sum += __shfl_xor(sum, 16);
      float w =
          attn_sn[(size_t)((bb * CH + h) * CLQ + q0 + srow) * CNBP + n] / sum;
      for (int t = st0; t < st0 + CNT / 2; ++t) S[srow][t] *= w;
    }
    __syncthreads();
    for (int kt = 0; kt < CNT / 32; ++kt) {
      v16bf a = load_a_f32(&S[0][0], CNT, 0, kt * 32);
#pragma unroll
      for (int nt = 0; nt < 4; ++nt) {
        v16bf b = load_bt_bf(Vp, CNT, nt * 16, kt * 32);
        ctx[nt] = wmma_bf(a, b, ctx[nt]);
      }
    }
    __syncthreads();
  }
  __bf16* Co = ctx_w_bf + (size_t)bb * CLQ * CDM + h * CDK;
  for (int nt = 0; nt < 4; ++nt)
    store_c_bf16(Co, CDM, q0, nt * 16, ctx[nt]);
}

// ---------------- host side ----------------
extern "C" void kernel_launch(void* const* d_in, const int* in_sizes, int n_in,
                              void* d_out, int out_size, void* d_ws, size_t ws_size,
                              hipStream_t stream) {
  const float* q    = (const float*)d_in[0];
  const float* k_s  = (const float*)d_in[1];
  const float* v_s  = (const float*)d_in[2];
  const float* k_w  = (const float*)d_in[3];
  const float* v_w  = (const float*)d_in[4];
  const float* bias_w = (const float*)d_in[5];
  const float* bias_s = (const float*)d_in[6];
  const float* topic  = (const float*)d_in[7];
  const float* tbias  = (const float*)d_in[8];
  const float* ptop   = (const float*)d_in[9];
  const float* ptbias = (const float*)d_in[10];
  const float* Wq = (const float*)d_in[11];
  const float* bq = (const float*)d_in[12];
  const float* Wk = (const float*)d_in[13];
  const float* bk = (const float*)d_in[14];
  const float* Wv = (const float*)d_in[15];
  const float* bv = (const float*)d_in[16];
  const float* W_fct = (const float*)d_in[17];
  const float* b_fct = (const float*)d_in[18];
  const float* W_fcp = (const float*)d_in[19];
  const float* b_fcp = (const float*)d_in[20];
  const float* W_fcs = (const float*)d_in[21];
  const float* b_fcs = (const float*)d_in[22];
  const float* W_fc  = (const float*)d_in[23];
  const float* b_fc  = (const float*)d_in[24];
  float* out = (float*)d_out;

  size_t off = 0;
  char* base = (char*)d_ws;
  auto allocb = [&](size_t elems) -> __bf16* {
    off = (off + 255) & ~(size_t)255;
    __bf16* p = (__bf16*)(base + off);
    off += elems * sizeof(__bf16);
    return p;
  };
  auto allocf = [&](size_t elems) -> float* {
    off = (off + 255) & ~(size_t)255;
    float* p = (float*)(base + off);
    off += elems * sizeof(float);
    return p;
  };

  const size_t nQ   = (size_t)CB * CLQ * CDM;
  const size_t nTp  = (size_t)CB * CTP * CDM;
  const size_t nPp  = (size_t)CB * CNBP * CDM;
  const size_t nW   = (size_t)CB * CNB * CNT * CDM;
  const size_t WS   = (size_t)CDM * CDM;

  __bf16* q_bf    = allocb(nQ);
  __bf16* topic_bf= allocb(nTp);
  __bf16* ptop_bf = allocb(nPp);
  __bf16* ks_bf   = allocb(nPp);
  __bf16* vs_bf   = allocb(nPp);
  __bf16* kw_bf   = allocb(nW);
  __bf16* vw_bf   = allocb(nW);
  __bf16* WqT = allocb(4 * WS);
  __bf16* WkT = allocb(4 * WS);
  __bf16* WvT = allocb(4 * WS);
  __bf16* WfctT = allocb(WS);
  __bf16* WfcpT = allocb(WS);
  __bf16* WfcsT = allocb(WS);
  __bf16* WfcT  = allocb(3 * WS);

  __bf16* q_t  = allocb(nQ);
  __bf16* q_sp = allocb(nQ);
  __bf16* q_wp = allocb(nQ);
  __bf16* k_t  = allocb(nTp);
  __bf16* vtT  = allocb(nTp);
  __bf16* q_pp = allocb(nTp);
  __bf16* k_pp = allocb(nPp);
  __bf16* k_spp = allocb(nPp);
  __bf16* vsppT = allocb(nPp);
  __bf16* kw_p = allocb(nW);
  __bf16* vwT  = allocb(nW);

  __bf16* attn_t_bf = allocb((size_t)CB * CH * CLQ * CTP);
  __bf16* apT       = allocb((size_t)CB * CH * CNBP * CTP);
  float*  attn_tp_f = allocf((size_t)CB * CH * CLQ * CNBP);
  float*  attn_sn_f = allocf((size_t)CB * CH * CLQ * CNBP);

  __bf16* ctx_t_bf = allocb(nQ);
  __bf16* ctx_s_bf = allocb(nQ);
  __bf16* ctx_w_bf = allocb(nQ);
  __bf16* concat_bf = allocb((size_t)CB * CLQ * 3 * CDM);

  auto conv = [&](const float* s, __bf16* d, size_t n) {
    k_conv<<<(int)((n + 255) / 256), 256, 0, stream>>>(s, d, (long)n);
  };
  auto convPad = [&](const float* s, __bf16* d, int srows, int drows, int cols,
                     int batch) {
    long total = (long)batch * drows * cols;
    k_conv_pad<<<(int)((total + 255) / 256), 256, 0, stream>>>(s, d, srows, drows,
                                                               cols, total);
  };
  auto convT = [&](const float* s, __bf16* d, int K, int N) {
    long total = (long)K * N;
    k_conv_T<<<(int)((total + 255) / 256), 256, 0, stream>>>(s, d, K, N);
  };
  auto gemm = [&](const __bf16* A, size_t sA, int lda, const __bf16* WT, size_t sW,
                  int ldwT, const float* bias, float* Cf, __bf16* Cb, __bf16* CbT,
                  size_t sC, int ldc, size_t sCT, int ldt, int M, int N, int K,
                  int batch) {
    int tiles = (M >> 5) * (N >> 5);
    dim3 g((tiles + 3) / 4, batch);
    k_gemm_bf16<<<g, 128, 0, stream>>>(A, sA, lda, WT, sW, ldwT, bias, Cf, Cb, CbT,
                                       sC, ldc, sCT, ldt, M, N, K);
  };

  // 1) conversions
  conv(q, q_bf, nQ);
  conv(k_w, kw_bf, nW);
  conv(v_w, vw_bf, nW);
  convPad(topic, topic_bf, CNTOP, CTP, CDM, CB);
  convPad(ptop, ptop_bf, CNB, CNBP, CDM, CB);
  convPad(k_s, ks_bf, CNB, CNBP, CDM, CB);
  convPad(v_s, vs_bf, CNB, CNBP, CDM, CB);
  for (int i = 0; i < 4; ++i) {
    convT(Wq + i * WS, WqT + i * WS, CDM, CDM);
    convT(Wk + i * WS, WkT + i * WS, CDM, CDM);
    convT(Wv + i * WS, WvT + i * WS, CDM, CDM);
  }
  convT(W_fct, WfctT, CDM, CDM);
  convT(W_fcp, WfcpT, CDM, CDM);
  convT(W_fcs, WfcsT, CDM, CDM);
  convT(W_fc, WfcT, 3 * CDM, CDM);

  const int M_Q = CB * CLQ;

  // 2) projections
  gemm(q_bf, 0, CDM, WqT + 0 * WS, 0, CDM, bq + 0 * CDM, nullptr, q_t, nullptr, 0, CDM, 0, 0, M_Q, CDM, CDM, 1);
  gemm(q_bf, 0, CDM, WqT + 2 * WS, 0, CDM, bq + 2 * CDM, nullptr, q_sp, nullptr, 0, CDM, 0, 0, M_Q, CDM, CDM, 1);
  gemm(q_bf, 0, CDM, WqT + 3 * WS, 0, CDM, bq + 3 * CDM, nullptr, q_wp, nullptr, 0, CDM, 0, 0, M_Q, CDM, CDM, 1);
  gemm(topic_bf, (size_t)CTP * CDM, CDM, WkT + 0 * WS, 0, CDM, bk + 0 * CDM, nullptr, k_t, nullptr, (size_t)CTP * CDM, CDM, 0, 0, CTP, CDM, CDM, CB);
  gemm(topic_bf, (size_t)CTP * CDM, CDM, WvT + 0 * WS, 0, CDM, bv + 0 * CDM, nullptr, nullptr, vtT, 0, 0, (size_t)CDM * CTP, CTP, CTP, CDM, CDM, CB);
  gemm(topic_bf, (size_t)CTP * CDM, CDM, WqT + 1 * WS, 0, CDM, bq + 1 * CDM, nullptr, q_pp, nullptr, (size_t)CTP * CDM, CDM, 0, 0, CTP, CDM, CDM, CB);
  gemm(ptop_bf, (size_t)CNBP * CDM, CDM, WkT + 1 * WS, 0, CDM, bk + 1 * CDM, nullptr, k_pp, nullptr, (size_t)CNBP * CDM, CDM, 0, 0, CNBP, CDM, CDM, CB);
  gemm(ks_bf, (size_t)CNBP * CDM, CDM, WkT + 2 * WS, 0, CDM, bk + 2 * CDM, nullptr, k_spp, nullptr, (size_t)CNBP * CDM, CDM, 0, 0, CNBP, CDM, CDM, CB);
  gemm(vs_bf, (size_t)CNBP * CDM, CDM, WvT + 2 * WS, 0, CDM, bv + 2 * CDM, nullptr, nullptr, vsppT, 0, 0, (size_t)CDM * CNBP, CNBP, CNBP, CDM, CDM, CB);
  gemm(kw_bf, (size_t)CNT * CDM, CDM, WkT + 3 * WS, 0, CDM, bk + 3 * CDM, nullptr, kw_p, nullptr, (size_t)CNT * CDM, CDM, 0, 0, CNT, CDM, CDM, CB * CNB);
  gemm(vw_bf, (size_t)CNT * CDM, CDM, WvT + 3 * WS, 0, CDM, bv + 3 * CDM, nullptr, nullptr, vwT, 0, 0, (size_t)CDM * CNT, CNT, CNT, CDM, CDM, CB * CNB);

  // 3) topic attention
  k_topic_attn<<<CB * CH * (CLQ / 16), 32, 0, stream>>>(q_t, k_t, vtT, tbias,
                                                        attn_t_bf, ctx_t_bf);
  // 4) topic->paragraph probs (transposed)
  k_para_attn<<<CB * CH * (CTP / 16), 32, 0, stream>>>(q_pp, k_pp, ptbias, apT);

  // 5) attn_tp = attn_t @ attn_p (batched over B*H)
  gemm(attn_t_bf, (size_t)CLQ * CTP, CTP, apT, (size_t)CNBP * CTP, CTP, nullptr,
       attn_tp_f, nullptr, nullptr, (size_t)CLQ * CNBP, CNBP, 0, 0, CLQ, CNBP,
       CTP, CB * CH);

  // 6) sentence attention
  k_sent_attn<<<CB * CH * (CLQ / 16), 32, 0, stream>>>(q_sp, k_spp, vsppT, bias_s,
                                                       attn_tp_f, attn_sn_f,
                                                       ctx_s_bf);
  // 7) word attention
  k_word_attn<<<CB * CH * (CLQ / 16), 32, 0, stream>>>(q_wp, kw_p, vwT, bias_w,
                                                       attn_sn_f, ctx_w_bf);

  // 8) per-branch fc -> concat [B*LQ, 3*DM]
  gemm(ctx_t_bf, 0, CDM, WfctT, 0, CDM, b_fct, nullptr, concat_bf + 0 * CDM, nullptr, 0, 3 * CDM, 0, 0, M_Q, CDM, CDM, 1);
  gemm(ctx_s_bf, 0, CDM, WfcpT, 0, CDM, b_fcp, nullptr, concat_bf + 1 * CDM, nullptr, 0, 3 * CDM, 0, 0, M_Q, CDM, CDM, 1);
  gemm(ctx_w_bf, 0, CDM, WfcsT, 0, CDM, b_fcs, nullptr, concat_bf + 2 * CDM, nullptr, 0, 3 * CDM, 0, 0, M_Q, CDM, CDM, 1);

  // 9) final fc -> d_out (fp32)
  gemm(concat_bf, 0, 3 * CDM, WfcT, 0, CDM, b_fc, out, nullptr, nullptr, 0, CDM,
       0, 0, M_Q, CDM, 3 * CDM, 1);
  (void)in_sizes; (void)n_in; (void)out_size; (void)ws_size;
}
